// TrueSlidingWindowMHA_3152505995884
// MI455X (gfx1250) — compile-verified, matrix operand order checked by reading, not run on hardware
//
#include <hip/hip_runtime.h>
#include <hip/hip_bf16.h>

// ---------------------------------------------------------------------------
// Windowed causal MHA for MI455X (gfx1250).
//   All matmuls: v_wmma_f32_16x16x32_bf16 (wave32).
//   All GEMM tile movement: global_load_async_to_lds_b128 (ASYNCcnt),
//   double-buffered, one barrier per K-step.
//
//   B=4, S=4096, D=1024, H=16, dh=64, WINDOW=128, nb=32
//   pre:      xbf = bf16(x); wqkvT = bf16(w_qkv)^T; woutT = bf16(w_out)^T
//   kernel 1: qkv  = xbf @ wqkvT^T          -> bf16 ws
//   kernel 2: block-diagonal causal softmax attention -> bf16 ws
//   kernel 3: out  = attn @ woutT^T         -> f32 d_out
// ---------------------------------------------------------------------------

typedef __attribute__((ext_vector_type(16))) __bf16 v16bf;
typedef __attribute__((ext_vector_type(8)))  __bf16 v8bf;
typedef __attribute__((ext_vector_type(8)))  float  v8f;

#define SEQ      4096
#define DMODEL   1024
#define NQKV     3072
#define HEADS    16
#define DH       64
#define WIN      128
#define NBLK     32
#define MTOT     16384          // B * SEQ

// LDS strides (elements), padded so every 16B granule stays 16B aligned.
#define GEMM_LDK 40             // 32 + 8
#define QK_STR   72             // 64 + 8
#define P_STR    136            // 128 + 8
#define T_STR    72             // transpose tile stride

// ---------------------------------------------------------------------------
// CDNA5 async copy: per-lane 16B global -> LDS, tracked by ASYNCcnt.
// ---------------------------------------------------------------------------
__device__ __forceinline__ void async_copy16(void* lds_dst, const void* gsrc) {
  asm volatile("global_load_async_to_lds_b128 %0, %1, off"
               :
               : "v"((unsigned)(size_t)lds_dst),
                 "v"((unsigned long long)(size_t)gsrc)
               : "memory");
}

__device__ __forceinline__ void wait_async0() {
  asm volatile("s_wait_asynccnt 0" ::: "memory");
}

__device__ __forceinline__ v8f wmma_bf16(v16bf a, v16bf b, v8f c) {
  return __builtin_amdgcn_wmma_f32_16x16x32_bf16(
      false, a, false, b, (short)0, c, false, false);
}

// 16x32 bf16 A-fragment (B-fragment identical from an N-major tile).
// Lanes 0-15: rows 0..15, K={0..7,16..23}; lanes 16-31: K={8..15,24..31}.
__device__ __forceinline__ v16bf load_frag(const __bf16* base, int stride) {
  const int lane = threadIdx.x & 31;
  const int r    = lane & 15;
  const int kh   = (lane >> 4) << 3;
  const __bf16* p = base + r * stride + kh;
  v8bf lo = *(const v8bf*)(p);
  v8bf hi = *(const v8bf*)(p + 16);
  return __builtin_shufflevector(lo, hi, 0, 1, 2, 3, 4, 5, 6, 7,
                                         8, 9, 10, 11, 12, 13, 14, 15);
}

// ---------------------------------------------------------------------------
// Pre-pass 1: elementwise f32 -> bf16 (8 elements / thread)
// ---------------------------------------------------------------------------
__global__ __launch_bounds__(256) void cvt_bf16_kernel(
    const float* __restrict__ in, __bf16* __restrict__ out) {
  const size_t i = ((size_t)blockIdx.x * 256 + threadIdx.x) * 8;
  float4 f0 = *(const float4*)(in + i);
  float4 f1 = *(const float4*)(in + i + 4);
  v8bf o;
  o[0] = (__bf16)f0.x; o[1] = (__bf16)f0.y; o[2] = (__bf16)f0.z; o[3] = (__bf16)f0.w;
  o[4] = (__bf16)f1.x; o[5] = (__bf16)f1.y; o[6] = (__bf16)f1.z; o[7] = (__bf16)f1.w;
  *(v8bf*)(out + i) = o;
}

// ---------------------------------------------------------------------------
// Pre-pass 2: transpose + convert  in (K x N f32) -> outT (N x K bf16)
// 64x64 tiles through LDS.  grid = (N/64, K/64)
// ---------------------------------------------------------------------------
__global__ __launch_bounds__(256) void transpose_cvt_kernel(
    const float* __restrict__ in, __bf16* __restrict__ outT, int K, int N) {
  __shared__ __bf16 T[64 * T_STR];
  const int n0 = blockIdx.x * 64;
  const int k0 = blockIdx.y * 64;
  const int tid = threadIdx.x;
#pragma unroll
  for (int t = 0; t < 4; ++t) {
    int idx = tid + 256 * t;                 // 1024 float4 granules
    int k = idx >> 4, c4 = idx & 15;
    float4 f = *(const float4*)(in + (size_t)(k0 + k) * N + n0 + c4 * 4);
    T[(c4 * 4 + 0) * T_STR + k] = (__bf16)f.x;
    T[(c4 * 4 + 1) * T_STR + k] = (__bf16)f.y;
    T[(c4 * 4 + 2) * T_STR + k] = (__bf16)f.z;
    T[(c4 * 4 + 3) * T_STR + k] = (__bf16)f.w;
  }
  __syncthreads();
#pragma unroll
  for (int t = 0; t < 2; ++t) {
    int idx = tid + 256 * t;                 // 512 v8bf granules
    int n = idx >> 3, c = idx & 7;
    v8bf v = *(const v8bf*)(&T[n * T_STR + c * 8]);
    *(v8bf*)(outT + (size_t)(n0 + n) * K + k0 + c * 8) = v;
  }
}

// ---------------------------------------------------------------------------
// NT-GEMM: C(MxN) = A(Mx1024 bf16) * Bt(Nx1024 bf16)^T, K-step 32,
// 128x128 block tiles, double-buffered async-to-LDS.
// grid = (M/128, N/128), 256 threads (8 waves, each 32x64 of C).
// ---------------------------------------------------------------------------
template <typename OutT>
__global__ __launch_bounds__(256) void gemm_nt_kernel(
    const __bf16* __restrict__ A, const __bf16* __restrict__ Bt,
    OutT* __restrict__ C, int ldc) {
  __shared__ __bf16 Asl[2][128 * GEMM_LDK];
  __shared__ __bf16 Bsl[2][128 * GEMM_LDK];

  const int m0   = blockIdx.x * 128;
  const int n0   = blockIdx.y * 128;
  const int tid  = threadIdx.x;
  const int wave = tid >> 5;
  const int mw   = (wave & 3) * 32;
  const int nw   = (wave >> 2) * 64;

  // Per-thread async granules: tile is 128 rows x 4 x 16B; 2 granules each.
  const int g0   = tid * 2;
  const int arow = g0 >> 2;
  const int ac0  = (g0 & 3) * 8;             // element offset of 1st granule

  v8f acc[2][4] = {};

  // prologue: issue tile 0 into buffer 0
#pragma unroll
  for (int i = 0; i < 2; ++i) {
    int c = ac0 + i * 8;
    async_copy16(&Asl[0][arow * GEMM_LDK + c], A  + (size_t)(m0 + arow) * DMODEL + c);
    async_copy16(&Bsl[0][arow * GEMM_LDK + c], Bt + (size_t)(n0 + arow) * DMODEL + c);
  }

  for (int k0 = 0; k0 < DMODEL; k0 += 32) {
    const int cur = (k0 >> 5) & 1;
    wait_async0();               // my tile-k granules are in LDS
    __syncthreads();             // everyone's are; previous compute done
    if (k0 + 32 < DMODEL) {      // overlap next tile copy with compute
      const int nxt = cur ^ 1;
#pragma unroll
      for (int i = 0; i < 2; ++i) {
        int c = ac0 + i * 8;
        async_copy16(&Asl[nxt][arow * GEMM_LDK + c],
                     A + (size_t)(m0 + arow) * DMODEL + k0 + 32 + c);
        async_copy16(&Bsl[nxt][arow * GEMM_LDK + c],
                     Bt + (size_t)(n0 + arow) * DMODEL + k0 + 32 + c);
      }
    }
    v16bf a0 = load_frag(&Asl[cur][(mw + 0) * GEMM_LDK], GEMM_LDK);
    v16bf a1 = load_frag(&Asl[cur][(mw + 16) * GEMM_LDK], GEMM_LDK);
#pragma unroll
    for (int nt = 0; nt < 4; ++nt) {
      v16bf b = load_frag(&Bsl[cur][(nw + nt * 16) * GEMM_LDK], GEMM_LDK);
      acc[0][nt] = wmma_bf16(a0, b, acc[0][nt]);
      acc[1][nt] = wmma_bf16(a1, b, acc[1][nt]);
    }
  }

  // C layout: lane%16 = N, vgpr r -> M = r + 8*(lane/16)
  const int lane = tid & 31;
  const int cn   = lane & 15;
  const int mr   = (lane >> 4) * 8;
#pragma unroll
  for (int mt = 0; mt < 2; ++mt)
#pragma unroll
    for (int nt = 0; nt < 4; ++nt) {
      int col = n0 + nw + nt * 16 + cn;
#pragma unroll
      for (int r = 0; r < 8; ++r) {
        int row = m0 + mw + mt * 16 + mr + r;
        C[(size_t)row * ldc + col] = (OutT)acc[mt][nt][r];
      }
    }
}

// ---------------------------------------------------------------------------
// Block-diagonal causal attention.  One workgroup per (b, h, window).
// ---------------------------------------------------------------------------
__global__ __launch_bounds__(256) void attn_kernel(
    const __bf16* __restrict__ qkv, __bf16* __restrict__ attn_out) {
  // Q+K region reused for P after the score phase.
  __shared__ __align__(16) char smem[2 * 128 * QK_STR * 2 + 64 * P_STR * 2];
  __bf16* Qs = (__bf16*)smem;                           // 128 x QK_STR
  __bf16* Ks = Qs + 128 * QK_STR;                       // 128 x QK_STR
  __bf16* Vt = (__bf16*)(smem + 2 * 128 * QK_STR * 2);  // 64 x P_STR (dh-major)
  __bf16* Ps = (__bf16*)smem;                           // 128 x P_STR (reuse)

  const int blk  = blockIdx.x;                          // ((b*H + h)*NBLK + win)
  const int win  = blk % NBLK;
  const int bh   = blk / NBLK;
  const int h    = bh % HEADS;
  const int b    = bh / HEADS;
  const int tok0 = b * SEQ + win * WIN;
  const int colq = h * DH;
  const int tid  = threadIdx.x;

  // Q/K rows: 128 x 64 bf16 = 128 x 8 granules -> async copies.
  // V: manual load + transposed scatter (dh-major) for the B fragment.
#pragma unroll
  for (int t = 0; t < 4; ++t) {
    int idx = tid + 256 * t;
    int row = idx >> 3, c = (idx & 7) * 8;
    size_t g = (size_t)(tok0 + row) * NQKV + colq + c;
    async_copy16(&Qs[row * QK_STR + c], qkv + g);
    async_copy16(&Ks[row * QK_STR + c], qkv + g + DMODEL);
    v8bf v = *(const v8bf*)(qkv + g + 2 * DMODEL);
#pragma unroll
    for (int j = 0; j < 8; ++j) Vt[(c + j) * P_STR + row] = v[j];
  }
  wait_async0();
  __syncthreads();

  const int wave = tid >> 5;       // query strip [16*wave, 16*wave+16)
  const int lane = tid & 31;
  const int ln   = lane & 15;      // key-within-tile (C-layout N)
  const int hi   = lane >> 4;      // row-half select

  // S = Q K^T : 16 queries x 128 keys per wave, dh = 64 (2 K-steps)
  v8f sc[8] = {};
  v16bf qa0 = load_frag(Qs + (wave * 16) * QK_STR + 0,  QK_STR);
  v16bf qa1 = load_frag(Qs + (wave * 16) * QK_STR + 32, QK_STR);
#pragma unroll
  for (int t = 0; t < 8; ++t) {
    v16bf kb0 = load_frag(Ks + (t * 16) * QK_STR + 0,  QK_STR);
    v16bf kb1 = load_frag(Ks + (t * 16) * QK_STR + 32, QK_STR);
    sc[t] = wmma_bf16(qa0, kb0, sc[t]);
    sc[t] = wmma_bf16(qa1, kb1, sc[t]);
  }

  // Causal mask + softmax.  Row m = 16*wave + 8*hi + r lives on the 16 lanes
  // of one half-wave; reduce with intra-half shfl_xor (masks 1,2,4,8).
  const float scale = 0.125f;      // dh^-0.5
#pragma unroll
  for (int r = 0; r < 8; ++r) {
    int q_idx = wave * 16 + hi * 8 + r;
    float mx = -3.0e38f;
#pragma unroll
    for (int t = 0; t < 8; ++t) {
      int k_idx = t * 16 + ln;
      float s = sc[t][r] * scale;
      s = (k_idx <= q_idx) ? s : -1.0e9f;
      sc[t][r] = s;
      mx = fmaxf(mx, s);
    }
    mx = fmaxf(mx, __shfl_xor(mx, 1, 32));
    mx = fmaxf(mx, __shfl_xor(mx, 2, 32));
    mx = fmaxf(mx, __shfl_xor(mx, 4, 32));
    mx = fmaxf(mx, __shfl_xor(mx, 8, 32));
    float sum = 0.0f;
#pragma unroll
    for (int t = 0; t < 8; ++t) {
      float p = __expf(sc[t][r] - mx);
      sc[t][r] = p;
      sum += p;
    }
    sum += __shfl_xor(sum, 1, 32);
    sum += __shfl_xor(sum, 2, 32);
    sum += __shfl_xor(sum, 4, 32);
    sum += __shfl_xor(sum, 8, 32);
    float inv = 1.0f / sum;
#pragma unroll
    for (int t = 0; t < 8; ++t) sc[t][r] *= inv;
  }

  __syncthreads();                 // done reading Q/K before overlaying with P
#pragma unroll
  for (int r = 0; r < 8; ++r) {
    int m = wave * 16 + hi * 8 + r;
#pragma unroll
    for (int t = 0; t < 8; ++t)
      Ps[m * P_STR + t * 16 + ln] = (__bf16)sc[t][r];
  }
  __syncthreads();

  // O = P @ V : 16 queries x 64 dh, K = 128 keys (4 K-steps)
  v8f o[4] = {};
#pragma unroll
  for (int ks = 0; ks < 4; ++ks) {
    v16bf pa = load_frag(Ps + (wave * 16) * P_STR + ks * 32, P_STR);
#pragma unroll
    for (int nt = 0; nt < 4; ++nt) {
      v16bf vb = load_frag(Vt + (nt * 16) * P_STR + ks * 32, P_STR);
      o[nt] = wmma_bf16(pa, vb, o[nt]);
    }
  }

#pragma unroll
  for (int nt = 0; nt < 4; ++nt) {
    int col = colq + nt * 16 + ln;
#pragma unroll
    for (int r = 0; r < 8; ++r) {
      int row = tok0 + wave * 16 + hi * 8 + r;
      attn_out[(size_t)row * DMODEL + col] = (__bf16)o[nt][r];
    }
  }
}

// ---------------------------------------------------------------------------
extern "C" void kernel_launch(void* const* d_in, const int* in_sizes, int n_in,
                              void* d_out, int out_size, void* d_ws,
                              size_t ws_size, hipStream_t stream) {
  (void)in_sizes; (void)n_in; (void)out_size; (void)ws_size;

  const float* x     = (const float*)d_in[0];   // (4,4096,1024)
  const float* w_qkv = (const float*)d_in[1];   // (1024,3072)
  const float* w_out = (const float*)d_in[2];   // (1024,1024)
  float*       out   = (float*)d_out;           // (4,4096,1024)

  __bf16* qkv_ws  = (__bf16*)d_ws;                              // 16384x3072
  __bf16* attn_ws = qkv_ws  + (size_t)MTOT * NQKV;              // 16384x1024
  __bf16* xbf     = attn_ws + (size_t)MTOT * DMODEL;            // 16384x1024
  __bf16* wqkvT   = xbf     + (size_t)MTOT * DMODEL;            // 3072x1024
  __bf16* woutT   = wqkvT   + (size_t)NQKV * DMODEL;            // 1024x1024

  cvt_bf16_kernel<<<(size_t)MTOT * DMODEL / (256 * 8), 256, 0, stream>>>(x, xbf);
  transpose_cvt_kernel<<<dim3(NQKV / 64, DMODEL / 64), 256, 0, stream>>>(
      w_qkv, wqkvT, DMODEL, NQKV);
  transpose_cvt_kernel<<<dim3(DMODEL / 64, DMODEL / 64), 256, 0, stream>>>(
      w_out, woutT, DMODEL, DMODEL);

  gemm_nt_kernel<__bf16><<<dim3(MTOT / 128, NQKV / 128), 256, 0, stream>>>(
      xbf, wqkvT, qkv_ws, NQKV);
  attn_kernel<<<4 * HEADS * NBLK, 256, 0, stream>>>(qkv_ws, attn_ws);
  gemm_nt_kernel<float><<<dim3(MTOT / 128, DMODEL / 128), 256, 0, stream>>>(
      attn_ws, woutT, out, DMODEL);
}
